// Embed_30872224923825
// MI455X (gfx1250) — matrix operand order repaired and checked by reading, not compiled, not run on hardware
//
#include <hip/hip_runtime.h>
#include <hip/hip_bf16.h>
#include <math.h>
#include <stdint.h>

typedef __attribute__((ext_vector_type(16))) _Float16 v16h;
typedef __attribute__((ext_vector_type(8)))  _Float16 v8h;
typedef __attribute__((ext_vector_type(8)))  float    v8f;
typedef __attribute__((ext_vector_type(4)))  unsigned int v4u;
typedef __attribute__((ext_vector_type(4)))  int      v4i;
typedef __attribute__((ext_vector_type(8)))  int      v8i;

#define HN 12
#define BN 64
#define LN 64
#define EN 768
#define DK 64
#define BLN 4096
#define TEN 384   /* 2L*3 */

// ---------------------------------------------------------------------------
// WMMA helpers: C(16x16,f32) += A(16x32 f16, row-major) x Bt(16x32 f16, row-major)^T
// Fragment layouts per CDNA5 ISA 7.12.2 (wave32).
// ---------------------------------------------------------------------------
__device__ __forceinline__ v16h ldfragA(const _Float16* A, int lda, int lane) {
    int r = lane & 15, hh = lane >> 4;
    const _Float16* p0 = A + r * lda + hh * 8;   // K = 8*hh + [0..7]
    const _Float16* p1 = p0 + 16;                // K = 16 + 8*hh + [0..7]
    v8h lo = *(const v8h*)p0;
    v8h hi = *(const v8h*)p1;
    v16h f;
#pragma unroll
    for (int i = 0; i < 8; i++) { f[i] = lo[i]; f[i + 8] = hi[i]; }
    return f;
}

__device__ __forceinline__ v16h ldfragB(const _Float16* Bt, int ldb, int lane) {
    int r = lane & 15, hh = lane >> 4;
    const _Float16* p0 = Bt + r * ldb + hh * 8;  // K = 8*hh + [0..7], 16 + 8*hh + [0..7]
    v8h lo = *(const v8h*)p0;
    v8h hi = *(const v8h*)(p0 + 16);
    v16h f;
#pragma unroll
    for (int i = 0; i < 8; i++) { f[i] = lo[i]; f[i + 8] = hi[i]; }
    return f;
}

__device__ __forceinline__ v8f wmma32(v16h a, v16h b, v8f c) {
    return __builtin_amdgcn_wmma_f32_16x16x32_f16(false, a, false, b, (short)0, c, false, false);
}

// ---------------------------------------------------------------------------
// Tensor Data Mover: DMA one 64x32 f16 tile (row stride 768 elems) into LDS.
// D# packing per CDNA5 ISA 8.3/8.4:
//   group0: count=1 | lds_addr[63:32] | global_addr[120:64] | type=2 @ [127:126]
//   group1: data_size=2B @ [17:16]; tensor_dim0=768 @ [79:48];
//           tensor_dim1=rows @ [111:80]; tile_dim0=32 @ [127:112];
//           tile_dim1=64 @ [143:128]; tensor_dim0_stride=768 @ [207:160]
// Issued by one wave; tracked by TENSORcnt.
// ---------------------------------------------------------------------------
__device__ __forceinline__ void tdm_load_tile64x32(const _Float16* gtile, unsigned ldsByteOff,
                                                   unsigned tensorRows) {
    unsigned long long ga = (unsigned long long)(uintptr_t)gtile;
    v4u g0;
    g0[0] = 1u;                                        // count=1, user descriptor
    g0[1] = ldsByteOff;                                // lds_addr
    g0[2] = (unsigned)ga;                              // global_addr[31:0]
    g0[3] = (unsigned)((ga >> 32) & 0x01FFFFFFull) | (2u << 30);  // addr[56:32] | type=2
    v8i g1;
    g1[0] = 0x00010000;                                // data_size = 2 bytes
    g1[1] = (int)((768u & 0xFFFFu) << 16);             // tensor_dim0 lo16
    g1[2] = (int)((768u >> 16) | ((tensorRows & 0xFFFFu) << 16)); // dim0 hi | dim1 lo
    g1[3] = (int)(((tensorRows >> 16) & 0xFFFFu) | (32u << 16));  // dim1 hi | tile_dim0=32
    g1[4] = 64;                                        // tile_dim1=64, tile_dim2=0
    g1[5] = 768;                                       // tensor_dim0_stride lo32
    g1[6] = 0;                                         // stride hi | dim1_stride lo
    g1[7] = 0;
    v4i zz = {0, 0, 0, 0};
#if __clang_major__ >= 23
    v8i z8 = {0, 0, 0, 0, 0, 0, 0, 0};
    __builtin_amdgcn_tensor_load_to_lds(g0, g1, zz, zz, z8, 0);
#else
    __builtin_amdgcn_tensor_load_to_lds(g0, g1, zz, zz, 0);
#endif
}

// ---------------------------------------------------------------------------
// f32 -> f16 convert
// ---------------------------------------------------------------------------
__global__ void cvt_kernel(const float* __restrict__ src, _Float16* __restrict__ dst, int n) {
    int i = blockIdx.x * 256 + threadIdx.x;
    if (i < n) dst[i] = (_Float16)src[i];
}

// transpose-gather relation_msg for edges (0,4,5): rmT[e][h][f][d] = rm[E(e)][h][d][f]
__global__ void rmt_kernel(const float* __restrict__ rm, _Float16* __restrict__ rmT) {
    int bid = blockIdx.x;              // 0..35 = e*12+h
    int e = bid / 12, h = bid % 12;
    int ee = (e == 0) ? 0 : (e + 3);   // {0,4,5}
    const float* src = rm + (size_t)(ee * 12 + h) * 4096;
    _Float16* dst = rmT + (size_t)bid * 4096;
    for (int idx = threadIdx.x; idx < 4096; idx += 256) {
        int d = idx >> 6, f = idx & 63;
        dst[f * 64 + d] = (_Float16)src[d * 64 + f];
    }
}

// ---------------------------------------------------------------------------
// GEMM: out = X[4096x768] * W[768x768]^T + bias. TDM double-buffered tiles.
// Head-major remapped f16 out:
//   dst[((head*64 + (m>>6)) * Rr + (m&63) + roff)*64 + (n&63)]
// grid (64,12), block 128 (4 waves), block tile 64x64.
// ---------------------------------------------------------------------------
__global__ void gemm_qkv(const _Float16* __restrict__ X, const _Float16* __restrict__ W,
                         const float* __restrict__ bias, _Float16* __restrict__ out,
                         int Rr, int roff) {
    __shared__ _Float16 ldsA[2][64 * 32];
    __shared__ _Float16 ldsB[2][64 * 32];
    const int lane = threadIdx.x & 31;
    const int w = threadIdx.x >> 5;
    const int m0 = blockIdx.x * 64;
    const int n0 = blockIdx.y * 64;
    v8f acc[4] = {};
    const unsigned aOff = (unsigned)(uintptr_t)&ldsA[0][0];  // LDS byte offset
    const unsigned bOff = (unsigned)(uintptr_t)&ldsB[0][0];
    const unsigned bufB = 64 * 32 * 2;                       // bytes per buffer

    if (w == 0) {
        tdm_load_tile64x32(X + (size_t)m0 * 768, aOff, 4096);
        tdm_load_tile64x32(W + (size_t)n0 * 768, bOff, 768);
    }
    for (int ks = 0; ks < 24; ks++) {
        int cur = ks & 1;
        if (w == 0) {
            if (ks + 1 < 24) {
                int nxt = cur ^ 1;
                tdm_load_tile64x32(X + (size_t)m0 * 768 + (ks + 1) * 32, aOff + nxt * bufB, 4096);
                tdm_load_tile64x32(W + (size_t)n0 * 768 + (ks + 1) * 32, bOff + nxt * bufB, 768);
                __builtin_amdgcn_s_wait_tensorcnt(2);  // current pair done; next pair in flight
            } else {
                __builtin_amdgcn_s_wait_tensorcnt(0);
            }
        }
        __syncthreads();
        v16h a = ldfragA(&ldsA[cur][0] + w * 16 * 32, 32, lane);
#pragma unroll
        for (int tn = 0; tn < 4; tn++) {
            v16h bf = ldfragB(&ldsB[cur][0] + tn * 16 * 32, 32, lane);
            acc[tn] = wmma32(a, bf, acc[tn]);
        }
        __syncthreads();   // readers done before TDM overwrites this buffer next step
    }
    int r = lane & 15, hh = lane >> 4;
#pragma unroll
    for (int tn = 0; tn < 4; tn++) {
#pragma unroll
        for (int v = 0; v < 8; v++) {
            int m = m0 + w * 16 + v + 8 * hh;
            int n = n0 + tn * 16 + r;
            float val = acc[tn][v] + bias[n];
            int head = n >> 6, d = n & 63;
            size_t dst = ((size_t)(head * 64 + (m >> 6)) * Rr + (m & 63) + roff) * 64 + d;
            out[dst] = (_Float16)val;
        }
    }
}

// ---------------------------------------------------------------------------
// qra[e][h][b] = q4[h][b] (64x64) x ra[e][h]^T (64x64).  grid 3*12*64, block 128.
// ---------------------------------------------------------------------------
__global__ void qra_kernel(const _Float16* __restrict__ q4, const _Float16* __restrict__ raH,
                           _Float16* __restrict__ qraH) {
    __shared__ _Float16 ldsQ[64 * 64];
    __shared__ _Float16 ldsR[64 * 64];
    int bid = blockIdx.x;
    int e = bid / 768, h = (bid / 64) % 12, b = bid % 64;
    const _Float16* Q = q4 + (size_t)(h * 64 + b) * 4096;
    const _Float16* R = raH + (size_t)(e * 12 + h) * 4096;
    for (int c = threadIdx.x; c < 512; c += 128) {
        ((uint4*)ldsQ)[c] = ((const uint4*)Q)[c];
        ((uint4*)ldsR)[c] = ((const uint4*)R)[c];
    }
    __syncthreads();
    const int lane = threadIdx.x & 31, w = threadIdx.x >> 5;
    _Float16* out = qraH + (size_t)((e * 12 + h) * 64 + b) * 4096;
    v8f acc[4] = {};
#pragma unroll
    for (int ks = 0; ks < 2; ks++) {
        v16h a = ldfragA(ldsQ + w * 16 * 64 + ks * 32, 64, lane);
#pragma unroll
        for (int tn = 0; tn < 4; tn++) {
            v16h bf = ldfragB(ldsR + tn * 16 * 64 + ks * 32, 64, lane);
            acc[tn] = wmma32(a, bf, acc[tn]);
        }
    }
    int r = lane & 15, hh = lane >> 4;
#pragma unroll
    for (int tn = 0; tn < 4; tn++)
#pragma unroll
        for (int v = 0; v < 8; v++)
            out[(w * 16 + v + 8 * hh) * 64 + tn * 16 + r] = (_Float16)acc[tn][v];
}

// ---------------------------------------------------------------------------
// Mask build: base edge masks, id-equality closure, repeat kill. One block per b.
// ---------------------------------------------------------------------------
__global__ void mask_kernel(const int* __restrict__ ids, const float* __restrict__ pre,
                            const float* __restrict__ back, const float* __restrict__ click,
                            unsigned char* __restrict__ maskB) {
    __shared__ int sid[64];
    __shared__ unsigned char sm[64 * TEN];
    __shared__ unsigned char srep[64];
    int b = blockIdx.x;
    for (int i = threadIdx.x; i < 64; i += 256) sid[i] = ids[b * 64 + i];
    __syncthreads();
    for (int idx = threadIdx.x; idx < 64 * TEN; idx += 256) {
        int l = idx / TEN, j = idx % TEN;
        int t = j / 3, e = j % 3;
        float mv = 0.f;
        size_t base = ((size_t)b * 64 + l) * 64;   // [B][L][L]
        if (e == 0)      { if (t >= 64) mv = click[base + (t - 64)]; }
        else if (e == 1) { if (t < 64)  mv = back[base + t]; }
        else             { if (t < 64)  mv = pre[base + t]; }
        sm[idx] = (mv > 0.f) ? 1 : 0;
    }
    if (threadIdx.x < 64) {
        int l = threadIdx.x; unsigned char rp = 0;
        for (int i = 0; i < l; i++) rp |= (sid[i] == sid[l]) ? 1 : 0;
        srep[l] = rp;
    }
    __syncthreads();
    for (int idx = threadIdx.x; idx < 64 * TEN; idx += 256) {
        int i = idx / TEN, j = idx % TEN;
        unsigned char v = 0;
        int idi = sid[i];
        for (int l = 0; l < 64; l++)
            v |= ((idi == sid[l]) ? sm[l * TEN + j] : (unsigned char)0);
        v = srep[i] ? 0 : v;
        maskB[(size_t)b * (64 * TEN) + idx] = v;
    }
}

// ---------------------------------------------------------------------------
// Fused attention per (h,b): scores -> masked softmax -> msg^T -> p*msg^T -> GELU
// block 256 (8 waves); all intermediates in LDS (~278 KB, CDNA5 WGP has 320 KB).
// ---------------------------------------------------------------------------
__global__ void attn_kernel(const _Float16* __restrict__ qraH, const _Float16* __restrict__ k4,
                            const _Float16* __restrict__ v4, const _Float16* __restrict__ rmTH,
                            const float* __restrict__ priP, const unsigned char* __restrict__ maskB,
                            _Float16* __restrict__ neighH) {
    extern __shared__ __align__(16) char smem[];
    _Float16* sQra = (_Float16*)smem;                   // 3*64*64
    _Float16* sK   = sQra + 3 * 4096;                   // 128*64
    _Float16* sV   = sK + 8192;                         // 128*64
    _Float16* sRmT = sV + 8192;                         // 3*64*64
    float*    sSc  = (float*)(sRmT + 3 * 4096);         // 64*384 f32
    _Float16* sP   = (_Float16*)(sSc + 64 * TEN);       // 64*384
    _Float16* sMT  = sP + 64 * TEN;                     // 64*384 (msg^T: [d][j])

    int h = blockIdx.x / 64, b = blockIdx.x % 64;
    const int tid = threadIdx.x, lane = tid & 31, w = tid >> 5;
    int r = lane & 15, hh = lane >> 4;

    for (int c = tid; c < 1536; c += 256) {
        int e = c >> 9, cc = c & 511;
        ((uint4*)sQra)[c] = ((const uint4*)qraH)[(size_t)((e * 12 + h) * 64 + b) * 512 + cc];
        ((uint4*)sRmT)[c] = ((const uint4*)rmTH)[(size_t)(e * 12 + h) * 512 + cc];
    }
    for (int c = tid; c < 1024; c += 256) {
        ((uint4*)sK)[c] = ((const uint4*)k4)[(size_t)(h * 64 + b) * 1024 + c];
        ((uint4*)sV)[c] = ((const uint4*)v4)[(size_t)(h * 64 + b) * 1024 + c];
    }
    __syncthreads();

    // scores: att[l][t*3+e] = (qra[e] @ k^T) * pri/sqrt(dk)
    for (int j = w; j < 96; j += 8) {
        int e = j >> 5, rem = j & 31, tm = rem >> 3, tn = rem & 7;
        v8f acc = {};
#pragma unroll
        for (int ks = 0; ks < 2; ks++) {
            v16h a  = ldfragA(sQra + e * 4096 + tm * 16 * 64 + ks * 32, 64, lane);
            v16h bf = ldfragB(sK + tn * 16 * 64 + ks * 32, 64, lane);
            acc = wmma32(a, bf, acc);
        }
        int ee = (e == 0) ? 0 : (e + 3);
        float scale = priP[ee * 12 + h] * 0.125f;  // sqrt(dk)=8
#pragma unroll
        for (int v = 0; v < 8; v++) {
            int l = tm * 16 + v + 8 * hh;
            int t = tn * 16 + r;
            sSc[l * TEN + t * 3 + e] = acc[v] * scale;
        }
    }
    // msg^T: sMT[d][t*3+e] = (v @ rmT[e]^T)[t][d]
    for (int j = w; j < 96; j += 8) {
        int e = j >> 5, rem = j & 31, tm = rem >> 2, tn = rem & 3;
        v8f acc = {};
#pragma unroll
        for (int ks = 0; ks < 2; ks++) {
            v16h a  = ldfragA(sV + tm * 16 * 64 + ks * 32, 64, lane);
            v16h bf = ldfragB(sRmT + e * 4096 + tn * 16 * 64 + ks * 32, 64, lane);
            acc = wmma32(a, bf, acc);
        }
#pragma unroll
        for (int v = 0; v < 8; v++) {
            int t = tm * 16 + v + 8 * hh;
            int d = tn * 16 + r;
            sMT[d * TEN + t * 3 + e] = (_Float16)acc[v];
        }
    }
    __syncthreads();

    // masked softmax: 8 rows per wave, 12 cols per lane
    const unsigned char* mrow = maskB + (size_t)b * (64 * TEN);
    for (int i = 0; i < 8; i++) {
        int l = w * 8 + i;
        float s[12]; unsigned char mk[12];
        float mx = -3.0e38f;
#pragma unroll
        for (int c = 0; c < 12; c++) {
            int j = lane + c * 32;
            mk[c] = mrow[l * TEN + j];
            float sv = mk[c] ? sSc[l * TEN + j] : -1.0e10f;
            s[c] = sv; mx = fmaxf(mx, sv);
        }
#pragma unroll
        for (int off = 16; off > 0; off >>= 1) mx = fmaxf(mx, __shfl_xor(mx, off));
        float sum = 0.f;
#pragma unroll
        for (int c = 0; c < 12; c++) { s[c] = expf(s[c] - mx); sum += s[c]; }
#pragma unroll
        for (int off = 16; off > 0; off >>= 1) sum += __shfl_xor(sum, off);
        float inv = 1.0f / sum;
#pragma unroll
        for (int c = 0; c < 12; c++) {
            int j = lane + c * 32;
            sP[l * TEN + j] = (_Float16)(mk[c] ? s[c] * inv : 0.f);
        }
    }
    __syncthreads();

    // neigh[l][d] = p[64x384] @ msgT^T, then exact GELU
    for (int j = w; j < 16; j += 8) {
        int tm = j >> 2, tn = j & 3;
        v8f acc = {};
#pragma unroll
        for (int ks = 0; ks < 12; ks++) {
            v16h a  = ldfragA(sP + tm * 16 * TEN + ks * 32, TEN, lane);
            v16h bf = ldfragB(sMT + tn * 16 * TEN + ks * 32, TEN, lane);
            acc = wmma32(a, bf, acc);
        }
#pragma unroll
        for (int v = 0; v < 8; v++) {
            int l = tm * 16 + v + 8 * hh;
            int d = tn * 16 + r;
            float x = acc[v];
            float g = 0.5f * x * (1.0f + erff(x * 0.70710678118654752f));
            neighH[(size_t)(b * 64 + l) * 768 + h * 64 + d] = (_Float16)g;
        }
    }
}

// ---------------------------------------------------------------------------
// Final: retype = neigh @ Wa^T + ba; blend with centers via sigmoid(skip[0]); LN.
// grid 256 (16 rows each), block 256 (8 waves).
// ---------------------------------------------------------------------------
__global__ void final_kernel(const _Float16* __restrict__ neighH, const _Float16* __restrict__ WaH,
                             const float* __restrict__ ba, const float* __restrict__ centers,
                             const float* __restrict__ skip, const float* __restrict__ gamma,
                             const float* __restrict__ beta, float* __restrict__ out) {
    extern __shared__ __align__(16) char smem[];
    _Float16* ldsA = (_Float16*)smem;          // 16*32
    _Float16* ldsB = ldsA + 16 * 32;           // 768*32
    float* ldsRes = (float*)(ldsB + 768 * 32); // 16*768 f32
    int m0 = blockIdx.x * 16;
    const int tid = threadIdx.x, lane = tid & 31, w = tid >> 5;
    v8f acc[6] = {};

    for (int k0 = 0; k0 < 768; k0 += 32) {
        __syncthreads();
        for (int idx = tid; idx < 512; idx += 256) {
            int row = idx >> 5, col = idx & 31;
            ldsA[idx] = neighH[(size_t)(m0 + row) * 768 + k0 + col];
        }
        for (int c = tid; c < 3072; c += 256) {
            int row = c >> 2, seg = (c & 3) * 8;
            ((uint4*)ldsB)[c] = *(const uint4*)(WaH + (size_t)row * 768 + k0 + seg);
        }
        __syncthreads();
        v16h a = ldfragA(ldsA, 32, lane);
#pragma unroll
        for (int i = 0; i < 6; i++) {
            int tn = w + 8 * i;
            v16h bf = ldfragB(ldsB + tn * 16 * 32, 32, lane);
            acc[i] = wmma32(a, bf, acc[i]);
        }
    }
    float alpha = 1.0f / (1.0f + expf(-skip[0]));
    int r = lane & 15, hh = lane >> 4;
#pragma unroll
    for (int i = 0; i < 6; i++) {
        int tn = w + 8 * i;
#pragma unroll
        for (int v = 0; v < 8; v++) {
            int m = v + 8 * hh;
            int n = tn * 16 + r;
            float x = acc[i][v] + ba[n];
            float res = x * alpha + centers[(size_t)(m0 + m) * 768 + n] * (1.0f - alpha);
            ldsRes[m * 768 + n] = res;
        }
    }
    __syncthreads();
    for (int i = 0; i < 2; i++) {
        int row = w * 2 + i;
        float s = 0.f, s2 = 0.f;
        for (int j = lane; j < 768; j += 32) {
            float x = ldsRes[row * 768 + j]; s += x; s2 += x * x;
        }
#pragma unroll
        for (int off = 16; off > 0; off >>= 1) {
            s += __shfl_xor(s, off); s2 += __shfl_xor(s2, off);
        }
        float mean = s * (1.0f / 768.0f);
        float var = s2 * (1.0f / 768.0f) - mean * mean;
        float rstd = rsqrtf(var + 1e-5f);
        for (int j = lane; j < 768; j += 32) {
            float x = ldsRes[row * 768 + j];
            out[(size_t)(m0 + row) * 768 + j] = gamma[j] * (x - mean) * rstd + beta[j];
        }
    }
}

// ---------------------------------------------------------------------------
extern "C" void kernel_launch(void* const* d_in, const int* in_sizes, int n_in,
                              void* d_out, int out_size, void* d_ws, size_t ws_size,
                              hipStream_t stream) {
    (void)in_sizes; (void)n_in; (void)out_size; (void)ws_size;
    const int*   center_ids = (const int*)d_in[0];
    const float* centers    = (const float*)d_in[1];
    const float* all_embs   = (const float*)d_in[2];
    const float* pre_mask   = (const float*)d_in[3];
    const float* back_mask  = (const float*)d_in[4];
    const float* click_mask = (const float*)d_in[5];
    const float* Wq = (const float*)d_in[7];
    const float* bq = (const float*)d_in[8];
    const float* Wk = (const float*)d_in[9];
    const float* bk = (const float*)d_in[10];
    const float* Wv = (const float*)d_in[11];
    const float* bv = (const float*)d_in[12];
    const float* Wa = (const float*)d_in[13];
    const float* ba = (const float*)d_in[14];
    const float* pri  = (const float*)d_in[15];
    const float* ratt = (const float*)d_in[16];
    const float* rmsg = (const float*)d_in[17];
    const float* skip  = (const float*)d_in[18];
    const float* gamma = (const float*)d_in[19];
    const float* beta  = (const float*)d_in[20];
    float* out = (float*)d_out;

    char* ws = (char*)d_ws;
    size_t off = 0;
    auto take = [&](size_t bytes) -> char* {
        char* p = ws + off;
        off += (bytes + 255) & ~(size_t)255;
        return p;
    };
    _Float16* cenH  = (_Float16*)take((size_t)BLN * EN * 2);
    _Float16* aeH   = (_Float16*)take((size_t)2 * BLN * EN * 2);
    _Float16* WqH   = (_Float16*)take((size_t)EN * EN * 2);
    _Float16* WkH   = (_Float16*)take((size_t)2 * EN * EN * 2);
    _Float16* WvH   = (_Float16*)take((size_t)2 * EN * EN * 2);
    _Float16* WaH   = (_Float16*)take((size_t)EN * EN * 2);
    _Float16* raH   = (_Float16*)take((size_t)3 * HN * DK * DK * 2);
    _Float16* rmTH  = (_Float16*)take((size_t)3 * HN * DK * DK * 2);
    _Float16* q4    = (_Float16*)take((size_t)HN * BN * LN * DK * 2);
    _Float16* k4    = (_Float16*)take((size_t)HN * BN * 2 * LN * DK * 2);
    _Float16* v4    = (_Float16*)take((size_t)HN * BN * 2 * LN * DK * 2);
    _Float16* qraH  = (_Float16*)take((size_t)3 * HN * BN * LN * DK * 2);
    unsigned char* maskB = (unsigned char*)take((size_t)BN * LN * TEN);
    _Float16* neighH = (_Float16*)take((size_t)BLN * EN * 2);

    const int EDG[3] = {0, 4, 5};

    // convert / gather
    cvt_kernel<<<(BLN * EN + 255) / 256, 256, 0, stream>>>(centers, cenH, BLN * EN);
    cvt_kernel<<<(2 * BLN * EN + 255) / 256, 256, 0, stream>>>(all_embs, aeH, 2 * BLN * EN);
    cvt_kernel<<<(EN * EN + 255) / 256, 256, 0, stream>>>(Wq, WqH, EN * EN);
    cvt_kernel<<<(2 * EN * EN + 255) / 256, 256, 0, stream>>>(Wk, WkH, 2 * EN * EN);
    cvt_kernel<<<(2 * EN * EN + 255) / 256, 256, 0, stream>>>(Wv, WvH, 2 * EN * EN);
    cvt_kernel<<<(EN * EN + 255) / 256, 256, 0, stream>>>(Wa, WaH, EN * EN);
    for (int e = 0; e < 3; e++)
        cvt_kernel<<<(HN * DK * DK + 255) / 256, 256, 0, stream>>>(
            ratt + (size_t)EDG[e] * HN * DK * DK, raH + (size_t)e * HN * DK * DK, HN * DK * DK);
    rmt_kernel<<<36, 256, 0, stream>>>(rmsg, rmTH);

    // QKV projections (type-selected weight halves), head-major outputs, TDM-fed
    dim3 gg(64, 12);
    gemm_qkv<<<gg, 128, 0, stream>>>(cenH, WqH, bq, q4, 64, 0);
    gemm_qkv<<<gg, 128, 0, stream>>>(aeH, WkH, bk, k4, 128, 0);
    gemm_qkv<<<gg, 128, 0, stream>>>(aeH + (size_t)BLN * EN, WkH + (size_t)EN * EN, bk + EN, k4, 128, 64);
    gemm_qkv<<<gg, 128, 0, stream>>>(aeH, WvH, bv, v4, 128, 0);
    gemm_qkv<<<gg, 128, 0, stream>>>(aeH + (size_t)BLN * EN, WvH + (size_t)EN * EN, bv + EN, v4, 128, 64);

    qra_kernel<<<3 * HN * BN, 128, 0, stream>>>(q4, raH, qraH);
    mask_kernel<<<BN, 256, 0, stream>>>(center_ids, pre_mask, back_mask, click_mask, maskB);

    size_t attnShm = (size_t)(3 * 4096 + 8192 + 8192 + 3 * 4096) * 2  // qra,k,v,rmT
                   + (size_t)64 * TEN * 4                              // scores f32
                   + (size_t)64 * TEN * 2 * 2;                         // p, msgT f16
    attn_kernel<<<HN * BN, 256, attnShm, stream>>>(qraH, k4, v4, rmTH, pri, maskB, neighH);

    size_t finShm = (size_t)16 * 32 * 2 + (size_t)768 * 32 * 2 + (size_t)16 * 768 * 4;
    final_kernel<<<256, 256, finShm, stream>>>(neighH, WaH, ba, centers, skip, gamma, beta, out);
}
